// GCNNDoubleQCritic_36498632081559
// MI455X (gfx1250) — matrix operand order repaired and checked by reading, not compiled, not run on hardware
//
#include <hip/hip_runtime.h>

// Problem constants (match reference)
#define BATCH      128
#define NPB        1024              // nodes per batch (N)
#define NODES      (BATCH * NPB)     // 131072
#define HDIM       128
#define KNN_K      5
#define OBS_PER    10
#define ACT_PER    2

typedef __attribute__((ext_vector_type(16))) _Float16 v16h;
typedef __attribute__((ext_vector_type(8)))  _Float16 v8h;
typedef __attribute__((ext_vector_type(4)))  _Float16 v4h;
typedef __attribute__((ext_vector_type(8)))  float    v8f;

// ---------------------------------------------------------------------------
// 1) Split (obs, action) -> robot_loc (f32, n x 2) and x padded to f16 n x 32
// ---------------------------------------------------------------------------
__global__ __launch_bounds__(256) void pack_input_kernel(
    const float* __restrict__ obs, const float* __restrict__ act,
    _Float16* __restrict__ xpad, float* __restrict__ loc, int n)
{
  const int i = blockIdx.x * 256 + threadIdx.x;
  if (i >= n) return;
  const float* o = obs + (size_t)i * OBS_PER;
  const float* a = act + (size_t)i * ACT_PER;
  loc[2 * i + 0] = o[0];
  loc[2 * i + 1] = o[1];
  _Float16* xp = xpad + (size_t)i * 32;
#pragma unroll
  for (int c = 0; c < 8; ++c) xp[c] = (_Float16)o[2 + c];
  xp[8] = (_Float16)a[0];
  xp[9] = (_Float16)a[1];
#pragma unroll
  for (int c = 10; c < 32; ++c) xp[c] = (_Float16)0.0f;
}

// ---------------------------------------------------------------------------
// 2) Brute-force 2D KNN per batch; 1024 locations staged in LDS.
// ---------------------------------------------------------------------------
__global__ __launch_bounds__(256) void knn_kernel(
    const float* __restrict__ loc, int* __restrict__ nbr, int n)
{
  __shared__ float2 sl[NPB];
  const int b    = blockIdx.x >> 2;     // batch
  const int seg  = blockIdx.x & 3;      // 4 blocks of 256 nodes per batch
  const int base = b * NPB;
  for (int j = threadIdx.x; j < NPB; j += 256)
    sl[j] = ((const float2*)loc)[base + j];
  __syncthreads();

  const int nb = seg * 256 + threadIdx.x;
  const float2 me = sl[nb];
  float bd[KNN_K];
  int   bi[KNN_K];
#pragma unroll
  for (int k = 0; k < KNN_K; ++k) { bd[k] = 3.4e38f; bi[k] = 0; }

  for (int j = 0; j < NPB; ++j) {
    if (j == nb) continue;
    const float dx = sl[j].x - me.x;
    const float dy = sl[j].y - me.y;
    const float d  = dx * dx + dy * dy;
    if (d < bd[KNN_K - 1]) {
      int p = KNN_K - 1;
      while (p > 0 && bd[p - 1] > d) { bd[p] = bd[p - 1]; bi[p] = bi[p - 1]; --p; }
      bd[p] = d; bi[p] = j;
    }
  }
  int* outp = nbr + (size_t)(base + nb) * 8;
#pragma unroll
  for (int k = 0; k < KNN_K; ++k) outp[k] = base + bi[k];
}

// ---------------------------------------------------------------------------
// 3) Pre-swizzle W[Kin x 128] (f32) into WMMA B-fragment layout (f16, K-padded).
//    Fragment linear index: ((kt*8 + nt)*32 + lane)*16 + half
//    B 32x16 layout: k = kt*32 + half + (lane>=16 ? 16 : 0), n = nt*16 + lane%16
// ---------------------------------------------------------------------------
__global__ __launch_bounds__(256) void pack_w_kernel(
    const float* __restrict__ W, _Float16* __restrict__ outp, int Kin, int KT)
{
  const int idx   = blockIdx.x * 256 + threadIdx.x;
  const int total = KT * 8 * 32 * 16;
  if (idx >= total) return;
  const int h    = idx & 15;
  const int lane = (idx >> 4) & 31;
  const int nt   = (idx >> 9) & 7;
  const int kt   = idx >> 12;
  const int k    = kt * 32 + h + ((lane >= 16) ? 16 : 0);
  const int nn   = nt * 16 + (lane & 15);
  const float v  = (k < Kin) ? W[(size_t)k * HDIM + nn] : 0.0f;
  outp[idx] = (_Float16)v;
}

// ---------------------------------------------------------------------------
// 4) GEMM: X[n x 128] (f32) = A[n x KT*32] (f16) @ W (packed fragments).
//    Block = 8 waves. Each wave-pair owns one M stream; wave "half" owns
//    n-tiles [4*half .. 4*half+3]. B fragments live in registers (loaded from
//    LDS once). Steady-state per m-tile: 2*KT global_load_b128 (A), 4*KT
//    v_wmma back-to-back, 32 b32 stores at constant offsets from ONE base
//    pointer (strength-reduced pointer increments across m-tiles).
// ---------------------------------------------------------------------------
template <int KT>
__global__ __launch_bounds__(256) void gcn_gemm_wmma(
    const _Float16* __restrict__ A, const _Float16* __restrict__ WP,
    float* __restrict__ X, int mt_per_wave)
{
  __shared__ __align__(32) _Float16 wlds[KT * 8 * 32 * 16];

  // Cooperative LDS fill of packed W fragments (16B chunks)
  {
    const int    cnt  = KT * 8 * 32 * 2;
    const uint4* wsrc = (const uint4*)WP;
    uint4*       wdst = (uint4*)wlds;
    for (int i = threadIdx.x; i < cnt; i += 256) wdst[i] = wsrc[i];
  }
  __syncthreads();

  const int wave  = threadIdx.x >> 5;
  const int lane  = threadIdx.x & 31;
  const int half  = wave & 1;          // n-tile group: 0 -> nt 0..3, 1 -> nt 4..7
  const int wpair = wave >> 1;         // 0..3: independent M streams

  // Load this wave's KT*4 B fragments into registers (one-time LDS reads)
  v16h bf[KT][4];
#pragma unroll
  for (int kt = 0; kt < KT; ++kt)
#pragma unroll
    for (int j = 0; j < 4; ++j)
      bf[kt][j] = *(const v16h*)(wlds + ((kt * 8 + half * 4 + j) * 32 + lane) * 16);

  const int kstride = KT * 32;
  const int koff    = (lane >= 16) ? 16 : 0;
  const int mrow    = lane & 15;
  const int ncol    = (half * 4) * 16 + (lane & 15);
  const int mtile0  = (blockIdx.x * 4 + wpair) * mt_per_wave;

  // Strength-reduced base pointers; all hot-loop offsets are immediates.
  const _Float16* arow = A + (size_t)(mtile0 * 16 + mrow) * kstride + koff;
  float*          xout = X + (size_t)(mtile0 * 16 + ((lane >= 16) ? 8 : 0)) * HDIM + ncol;

  for (int t = 0; t < mt_per_wave; ++t) {
    v8f acc[4] = {};
#pragma unroll
    for (int kt = 0; kt < KT; ++kt) {
      union { v16h v; v8h h[2]; } af;
      af.h[0] = *(const v8h*)(arow + kt * 32);        // 16B global load
      af.h[1] = *(const v8h*)(arow + kt * 32 + 8);    // 16B global load
#pragma unroll
      for (int j = 0; j < 4; ++j)
        acc[j] = __builtin_amdgcn_wmma_f32_16x16x32_f16(
            false, af.v, false, bf[kt][j], (short)0, acc[j], false, false);
    }
#pragma unroll
    for (int j = 0; j < 4; ++j)
#pragma unroll
      for (int v = 0; v < 8; ++v)
        xout[v * HDIM + j * 16] = acc[j][v];          // constant immediates

    arow += 16 * kstride;
    xout += 16 * HDIM;
  }
}

// ---------------------------------------------------------------------------
// 5) GCN aggregate: H[i] = relu((X[i] + sum_k X[nbr[i][k]]) / 6 + bias), f16 out.
//    deg == 6 for every node (each node is dst of exactly K=5 edges).
// ---------------------------------------------------------------------------
__global__ __launch_bounds__(256) void gcn_agg_relu(
    const float* __restrict__ X, const int* __restrict__ nbr,
    const float* __restrict__ bias, _Float16* __restrict__ H, int n)
{
  const int wave = threadIdx.x >> 5, lane = threadIdx.x & 31;
  const int node = blockIdx.x * 8 + wave;
  if (node >= n) return;
  const int c = lane << 2;

  float4 s = *(const float4*)(X + (size_t)node * HDIM + c);
  const int* nb = nbr + (size_t)node * 8;
#pragma unroll
  for (int k = 0; k < KNN_K; ++k) {
    const float4 t = *(const float4*)(X + (size_t)nb[k] * HDIM + c);
    s.x += t.x; s.y += t.y; s.z += t.z; s.w += t.w;
  }
  const float4 b4 = *(const float4*)(bias + c);
  const float inv6 = 1.0f / 6.0f;
  v4h o;
  o[0] = (_Float16)fmaxf(fmaf(s.x, inv6, b4.x), 0.0f);
  o[1] = (_Float16)fmaxf(fmaf(s.y, inv6, b4.y), 0.0f);
  o[2] = (_Float16)fmaxf(fmaf(s.z, inv6, b4.z), 0.0f);
  o[3] = (_Float16)fmaxf(fmaf(s.w, inv6, b4.w), 0.0f);
  *(v4h*)(H + (size_t)node * HDIM + c) = o;
}

// ---------------------------------------------------------------------------
// 6) Head: out[i] = dot(H[i], mlp_w) + mlp_b. Wave per node, shuffle reduce.
// ---------------------------------------------------------------------------
__global__ __launch_bounds__(256) void gcn_head(
    const _Float16* __restrict__ H, const float* __restrict__ w,
    const float* __restrict__ b, float* __restrict__ outp, int n)
{
  const int wave = threadIdx.x >> 5, lane = threadIdx.x & 31;
  const int node = blockIdx.x * 8 + wave;
  if (node >= n) return;
  const int c = lane << 2;
  const v4h    h4 = *(const v4h*)(H + (size_t)node * HDIM + c);
  const float4 w4 = *(const float4*)(w + c);
  float s = (float)h4[0] * w4.x + (float)h4[1] * w4.y +
            (float)h4[2] * w4.z + (float)h4[3] * w4.w;
#pragma unroll
  for (int off = 16; off > 0; off >>= 1) s += __shfl_xor(s, off, 32);
  if (lane == 0) outp[node] = s + b[0];
}

// ---------------------------------------------------------------------------
// Launch
// ---------------------------------------------------------------------------
extern "C" void kernel_launch(void* const* d_in, const int* in_sizes, int n_in,
                              void* d_out, int out_size, void* d_ws, size_t ws_size,
                              hipStream_t stream) {
  (void)in_sizes; (void)n_in; (void)out_size; (void)ws_size;
  const int n = NODES;

  const float* obs = (const float*)d_in[0];
  const float* act = (const float*)d_in[1];
  const float* W[6]  = {(const float*)d_in[2],  (const float*)d_in[4],  (const float*)d_in[6],
                        (const float*)d_in[8],  (const float*)d_in[10], (const float*)d_in[12]};
  const float* Bv[6] = {(const float*)d_in[3],  (const float*)d_in[5],  (const float*)d_in[7],
                        (const float*)d_in[9],  (const float*)d_in[11], (const float*)d_in[13]};
  const float* mlp_w = (const float*)d_in[14];
  const float* mlp_b = (const float*)d_in[15];
  float* outp = (float*)d_out;

  // Carve workspace (256B aligned slabs)
  char* p = (char*)d_ws;
  auto carve = [&](size_t bytes) -> char* {
    char* r = p;
    p += (bytes + 255) & ~(size_t)255;
    return r;
  };
  float*    loc  = (float*)   carve((size_t)n * 2 * sizeof(float));       //  1 MB
  _Float16* xpad = (_Float16*)carve((size_t)n * 32 * sizeof(_Float16));   //  8 MB
  int*      nbr  = (int*)     carve((size_t)n * 8 * sizeof(int));         //  4 MB
  float*    X    = (float*)   carve((size_t)n * HDIM * sizeof(float));    // 64 MB
  _Float16* Hbuf = (_Float16*)carve((size_t)n * HDIM * sizeof(_Float16)); // 32 MB
  _Float16* WP[6];
  for (int i = 0; i < 6; ++i)
    WP[i] = (_Float16*)carve((size_t)4 * 8 * 32 * 16 * sizeof(_Float16)); // 32 KB each

  // Preprocess
  pack_input_kernel<<<n / 256, 256, 0, stream>>>(obs, act, xpad, loc, n);
  knn_kernel<<<BATCH * (NPB / 256), 256, 0, stream>>>(loc, nbr, n);
  const int kts[6] = {1, 4, 4, 1, 4, 4};
  for (int i = 0; i < 6; ++i) {
    const int kin   = (i % 3 == 0) ? OBS_PER : HDIM;   // TRUNK=10 or 128
    const int total = kts[i] * 8 * 32 * 16;
    pack_w_kernel<<<(total + 255) / 256, 256, 0, stream>>>(W[i], WP[i], kin, kts[i]);
  }

  // GEMM decomposition: 8192 m-tiles; 4 M streams per block, MT tiles each.
  constexpr int MT = 8;                       // m-tiles per wave
  const int mtiles    = n / 16;               // 8192
  const int gemm_grid = mtiles / (4 * MT);    // 256 blocks (exact tiling)
  const int node_grid = n / 8;                // 8 nodes per block (wave each)

  for (int s = 0; s < 2; ++s) {
    const int wi = s * 3;
    gcn_gemm_wmma<1><<<gemm_grid, 256, 0, stream>>>(xpad, WP[wi + 0], X, MT);
    gcn_agg_relu    <<<node_grid, 256, 0, stream>>>(X, nbr, Bv[wi + 0], Hbuf, n);
    gcn_gemm_wmma<4><<<gemm_grid, 256, 0, stream>>>(Hbuf, WP[wi + 1], X, MT);
    gcn_agg_relu    <<<node_grid, 256, 0, stream>>>(X, nbr, Bv[wi + 1], Hbuf, n);
    gcn_gemm_wmma<4><<<gemm_grid, 256, 0, stream>>>(Hbuf, WP[wi + 2], X, MT);
    gcn_agg_relu    <<<node_grid, 256, 0, stream>>>(X, nbr, Bv[wi + 2], Hbuf, n);
    gcn_head        <<<node_grid, 256, 0, stream>>>(Hbuf, mlp_w, mlp_b, outp + (size_t)s * n, n);
  }
}
